// CWT__28716151341383
// MI455X (gfx1250) — matrix operand order repaired and checked by reading, not compiled, not run on hardware
//
#include <hip/hip_runtime.h>

// ---------------------------------------------------------------------------
// CWT (Morlet, pywt-compatible) on MI455X / gfx1250.
//
// out[n,t,c,f] = sum_d w_{s=f+1}[d] * x[n,t+d,c]   (zero-padded x)
// Implemented as f16 WMMA GEMM with fp32-accurate hi/lo splitting:
//   out ~= Ah*Bh + Ah*Bl + Al*Bh      (A = signal windows, B = filter taps)
// 8-phase trick: A rows stride 8 in t (so ds_load_b128 of A fragments are
// 16B aligned & conflict-free); phase p is absorbed into pre-shifted weight
// matrices Wp[k] = W[k-p], stored pre-swizzled in B-fragment order.
// ---------------------------------------------------------------------------

typedef _Float16 v16h __attribute__((ext_vector_type(16)));
typedef _Float16 v8h  __attribute__((ext_vector_type(8)));
typedef float    v8f  __attribute__((ext_vector_type(8)));

#define NB      64          // batch
#define TLEN    1024        // time
#define CCH     64          // channels
#define NF      30          // scales / filters
#define NFP     32          // padded filter dim (2 N-tiles of 16)
#define KTOT    512         // padded tap window (16 K-steps of 32)
#define KSTEPS  16
#define PHASES  8
#define PAD     241         // max left reach: d_min = -8*30-1 = -241
#define SIGLEN  1536        // 241 + 1024 + right pad, covers max idx 1527
#define KKMAX   481         // d_max - d_min = 240 + 241
#define WHALVES (PHASES*KSTEPS*2*32*16)   // 131072 f16 per (hi|lo) table
#define WBYTES  (WHALVES*2)               // 262144 bytes
#define OUTX    ((long)NB*TLEN*CCH*NF)    // 125829120 floats of x_, then y

// ---------------------------------------------------------------------------
// Kernel 1: int_psi = cumsum(exp(-x^2/2)*cos(5x)) * step  on 1024-pt grid.
// Single block, Hillis-Steele inclusive scan in LDS.
// ---------------------------------------------------------------------------
__global__ __launch_bounds__(1024) void cwt_psi_scan(float* __restrict__ int_psi) {
    __shared__ float buf[1024];
    const int i = threadIdx.x;
    const float xg = -8.0f + 16.0f * (float)i / 1023.0f;
    buf[i] = expf(-0.5f * xg * xg) * cosf(5.0f * xg);
    __syncthreads();
    for (int off = 1; off < 1024; off <<= 1) {
        float v = (i >= off) ? buf[i - off] : 0.0f;
        __syncthreads();
        buf[i] += v;
        __syncthreads();
    }
    int_psi[i] = buf[i] * (16.0f / 1023.0f);
}

// filt_s[m] = int_psi[ floor((16s-m)*1023/(16s)) ], zero outside [0,16s]
__device__ __forceinline__ float filt_val(int s, int m, const float* int_psi) {
    if (m < 0 || m > 16 * s) return 0.0f;
    const int j = ((16 * s - m) * 1023) / (16 * s);
    return int_psi[j];
}

// ---------------------------------------------------------------------------
// Kernel 2: build 8-phase, hi/lo-split weights, pre-swizzled in the WMMA
// B-fragment layout: block(p,ks,nt) of 32 lanes x 16 halves, lane&15 = filter
// column (n = nt*16 + lane%16), half i = K in table order for h = lane/16.
// Wp[k][f] = W[k-p][f],  W[kk][f] = w_{s=f+1}[kk - 241].
// ---------------------------------------------------------------------------
__global__ __launch_bounds__(256) void cwt_build_weights(const float* __restrict__ int_psi,
                                                         _Float16* __restrict__ Wh,
                                                         _Float16* __restrict__ Wl) {
    const int idx  = blockIdx.x * 256 + threadIdx.x;   // [0, 131072)
    const int i    = idx & 15;
    const int lane = (idx >> 4) & 31;
    const int nt   = (idx >> 9) & 1;
    const int ks   = (idx >> 10) & 15;
    const int p    = (idx >> 14) & 7;

    const int k  = ks * 32 + (i < 8 ? 8 * (lane >> 4) + i : 16 + 8 * (lane >> 4) + (i - 8));
    const int kk = k - p;
    const int f  = nt * 16 + (lane & 15);

    float wv = 0.0f;
    if (f < NF && kk >= 0 && kk <= KKMAX) {
        const int s = f + 1;
        const int d = kk - PAD;
        const float a = filt_val(s, 8 * s - d,     int_psi);
        const float b = filt_val(s, 8 * s - 1 - d, int_psi);
        wv = -sqrtf((float)s) * (a - b);
    }
    const _Float16 h = (_Float16)wv;
    Wh[idx] = h;
    Wl[idx] = (_Float16)(wv - (float)h);
}

// ---------------------------------------------------------------------------
// Kernel 3: main CWT. One workgroup (8 waves) per (n,c) signal.
// Wave w owns t-group t0 = 128*w; rows of each 16x512 A tile are t0+8*row.
// Per phase p: D[M][f] = sum_k A[M][k] * Wp[k][f]  ->  out[t0 + 8*M + p][f].
// ---------------------------------------------------------------------------
__global__ __launch_bounds__(256) void cwt_main(const float* __restrict__ x,
                                                const _Float16* __restrict__ Wh,
                                                const _Float16* __restrict__ Wl,
                                                float* __restrict__ out) {
    __shared__ __align__(16) _Float16 sigh[SIGLEN];
    __shared__ __align__(16) _Float16 sigl[SIGLEN];

    const int blk = blockIdx.x;
    const int n = blk >> 6;
    const int c = blk & 63;
    const int tid = threadIdx.x;

    // zero pad region, then deposit hi/lo split of the signal
    for (int i = tid; i < SIGLEN; i += 256) {
        sigh[i] = (_Float16)0.0f;
        sigl[i] = (_Float16)0.0f;
    }
    __syncthreads();
    for (int i = tid; i < TLEN; i += 256) {
        const float v = x[((long)n * TLEN + i) * CCH + c];   // strided; L2-resident
        const _Float16 h = (_Float16)v;
        sigh[PAD + i] = h;
        sigl[PAD + i] = (_Float16)(v - (float)h);
    }
    __syncthreads();

    const int wave = tid >> 5;
    const int lane = tid & 31;
    const int row  = lane & 15;     // A-tile row (M)
    const int hiL  = lane >> 4;     // lane group: K sub-block select
    const int t0   = wave * 128;

    // A fragment per-lane base (in halves): lanes<16 take K groups {0-7,16-23},
    // lanes>=16 take {8-15,24-31}  -> +8 halves. Always multiple of 8 -> 16B.
    const int aoff = t0 + 8 * row + 8 * hiL;

    for (int p = 0; p < PHASES; ++p) {
        v8f acc0 = {};
        v8f acc1 = {};
        const long wrow = ((long)p * KSTEPS) * (2 * 512) + (long)lane * 16;
        __builtin_prefetch(Wh + wrow, 0, 0);   // global_prefetch_b8 (current phase slice)
        __builtin_prefetch(Wl + wrow, 0, 0);

        for (int ks = 0; ks < KSTEPS; ++ks) {
            const int k0 = ks * 32;

            // ---- A fragments (signal hi/lo) from LDS: 2x ds_load_b128 each
            const v8h* ph = (const v8h*)(sigh + aoff + k0);
            const v8h* pl = (const v8h*)(sigl + aoff + k0);
            const v8h ah0 = ph[0], ah1 = ph[2];     // +0 and +16 halves
            const v8h al0 = pl[0], al1 = pl[2];
            const v16h ah = __builtin_shufflevector(ah0, ah1,
                0,1,2,3,4,5,6,7,8,9,10,11,12,13,14,15);
            const v16h al = __builtin_shufflevector(al0, al1,
                0,1,2,3,4,5,6,7,8,9,10,11,12,13,14,15);

            // ---- B fragments (weights hi/lo, both N-tiles): coalesced b128
            const long boff0 = wrow + (long)ks * (2 * 512);   // nt = 0
            const long boff1 = boff0 + 512;                   // nt = 1
            const v8h* b0hp = (const v8h*)(Wh + boff0);
            const v8h* b0lp = (const v8h*)(Wl + boff0);
            const v8h* b1hp = (const v8h*)(Wh + boff1);
            const v8h* b1lp = (const v8h*)(Wl + boff1);
            const v16h b0h = __builtin_shufflevector(b0hp[0], b0hp[1],
                0,1,2,3,4,5,6,7,8,9,10,11,12,13,14,15);
            const v16h b0l = __builtin_shufflevector(b0lp[0], b0lp[1],
                0,1,2,3,4,5,6,7,8,9,10,11,12,13,14,15);
            const v16h b1h = __builtin_shufflevector(b1hp[0], b1hp[1],
                0,1,2,3,4,5,6,7,8,9,10,11,12,13,14,15);
            const v16h b1l = __builtin_shufflevector(b1lp[0], b1lp[1],
                0,1,2,3,4,5,6,7,8,9,10,11,12,13,14,15);

            // ---- 6 WMMAs: (Ah*Bh + Ah*Bl + Al*Bh) for both N-tiles
            acc0 = __builtin_amdgcn_wmma_f32_16x16x32_f16(false, ah, false, b0h,
                                                          (short)0, acc0, false, false);
            acc0 = __builtin_amdgcn_wmma_f32_16x16x32_f16(false, ah, false, b0l,
                                                          (short)0, acc0, false, false);
            acc0 = __builtin_amdgcn_wmma_f32_16x16x32_f16(false, al, false, b0h,
                                                          (short)0, acc0, false, false);
            acc1 = __builtin_amdgcn_wmma_f32_16x16x32_f16(false, ah, false, b1h,
                                                          (short)0, acc1, false, false);
            acc1 = __builtin_amdgcn_wmma_f32_16x16x32_f16(false, ah, false, b1l,
                                                          (short)0, acc1, false, false);
            acc1 = __builtin_amdgcn_wmma_f32_16x16x32_f16(false, al, false, b1h,
                                                          (short)0, acc1, false, false);
        }

        // ---- store: D element (VGPR r, lane) = D[M = r + 8*hiL][f = lane&15]
        const int fc = lane & 15;
        #pragma unroll
        for (int r = 0; r < 8; ++r) {
            const int  M    = r + 8 * hiL;
            const long t    = (long)t0 + 8 * M + p;
            const long base = (((long)n * TLEN + t) * CCH + c) * NF;
            out[base + fc] = acc0[r];               // f = 0..15
            if (fc < NF - 16)                        // f = 16..29 (skip 30,31 pad)
                out[base + 16 + fc] = acc1[r];
        }
    }
}

// ---------------------------------------------------------------------------
// Kernel 4: append y (int64 labels) after the CWT block of the tuple output.
// ---------------------------------------------------------------------------
__global__ void cwt_copy_y(const long long* __restrict__ y, float* __restrict__ out) {
    const int i = threadIdx.x;
    if (i < NB) out[OUTX + i] = (float)y[i];
}

extern "C" void kernel_launch(void* const* d_in, const int* in_sizes, int n_in,
                              void* d_out, int out_size, void* d_ws, size_t ws_size,
                              hipStream_t stream) {
    (void)in_sizes; (void)n_in; (void)out_size; (void)ws_size;
    const float*     x = (const float*)d_in[0];
    const long long* y = (const long long*)d_in[1];
    float* out = (float*)d_out;

    // workspace: [int_psi 4KB][Wh 256KB][Wl 256KB]
    float*    int_psi = (float*)d_ws;
    _Float16* Wh = (_Float16*)((char*)d_ws + 4096);
    _Float16* Wl = (_Float16*)((char*)d_ws + 4096 + WBYTES);

    cwt_psi_scan     <<<1,    1024, 0, stream>>>(int_psi);
    cwt_build_weights<<<512,  256,  0, stream>>>(int_psi, Wh, Wl);
    cwt_main         <<<NB*CCH, 256, 0, stream>>>(x, Wh, Wl, out);
    cwt_copy_y       <<<1,    64,   0, stream>>>(y, out);
}
